// GCN_30305289241273
// MI455X (gfx1250) — compile-verified, matrix operand order checked by reading, not run on hardware
//
#include <hip/hip_runtime.h>
#include <hip/hip_bf16.h>

typedef __attribute__((ext_vector_type(16))) _Float16 v16h;
typedef __attribute__((ext_vector_type(8)))  _Float16 h8;
typedef __attribute__((ext_vector_type(8)))  float    v8f;

#define NNODES 50000
#define NEDGES 800000
#define INC 128
#define HC  128
#define OUTC 64

// ---------------- elementwise helpers ----------------

__global__ void f32_to_f16_kernel(const float* __restrict__ in, _Float16* __restrict__ out, int n) {
    int i = blockIdx.x * blockDim.x + threadIdx.x;
    if (i < n) out[i] = (_Float16)in[i];
}

__global__ void deg_init_kernel(float* __restrict__ deg, int n) {
    int i = blockIdx.x * blockDim.x + threadIdx.x;
    if (i < n) deg[i] = 1.0f;   // self-loop weight
}

__global__ void deg_accum_kernel(const int* __restrict__ dst, const float* __restrict__ ew,
                                 float* __restrict__ deg, int e) {
    int i = blockIdx.x * blockDim.x + threadIdx.x;
    if (i < e) atomicAdd(&deg[dst[i]], ew[i]);
}

__global__ void dis_kernel(const float* __restrict__ deg, float* __restrict__ dis, int n) {
    int i = blockIdx.x * blockDim.x + threadIdx.x;
    if (i < n) {
        float d = deg[i];
        dis[i] = (d > 0.0f) ? rsqrtf(fmaxf(d, 1e-12f)) : 0.0f;
    }
}

__global__ void norm_kernel(const int* __restrict__ src, const int* __restrict__ dst,
                            const float* __restrict__ ew, const float* __restrict__ dis,
                            float* __restrict__ nrm, int e) {
    int i = blockIdx.x * blockDim.x + threadIdx.x;
    if (i < e) nrm[i] = dis[src[i]] * ew[i] * dis[dst[i]];
}

// agg[i,c] = h[i,c] * dis[i]^2   (self-loop contribution; also serves as zero-init)
__global__ void self_init_kernel(const float* __restrict__ h, const float* __restrict__ dis,
                                 float* __restrict__ agg, int total, int cshift) {
    int i = blockIdx.x * blockDim.x + threadIdx.x;
    if (i < total) {
        float dv = dis[i >> cshift];
        agg[i] = h[i] * dv * dv;
    }
}

// agg[dst] += h[src] * norm : one thread per (edge, 4-feature chunk)
__global__ void edge_scatter_kernel(const float* __restrict__ h, const int* __restrict__ src,
                                    const int* __restrict__ dst, const float* __restrict__ nrm,
                                    float* __restrict__ agg, int e, int c, int chunkShift) {
    long long total = (long long)e << chunkShift;   // chunks of 4 features
    long long t = (long long)blockIdx.x * blockDim.x + threadIdx.x;
    if (t >= total) return;
    int ei = (int)(t >> chunkShift);
    int c4 = ((int)t & ((1 << chunkShift) - 1)) << 2;
    int s = src[ei], d = dst[ei];
    float w = nrm[ei];
    const float4 v = *(const float4*)(h + (size_t)s * c + c4);
    float* o = agg + (size_t)d * c + c4;
    atomicAdd(o + 0, v.x * w);
    atomicAdd(o + 1, v.y * w);
    atomicAdd(o + 2, v.z * w);
    atomicAdd(o + 3, v.w * w);
}

// out_f16 = relu( bn( agg + bias ) )
__global__ void bias_bn_relu_kernel(const float* __restrict__ agg, const float* __restrict__ b,
                                    const float* __restrict__ g, const float* __restrict__ be,
                                    const float* __restrict__ m, const float* __restrict__ v,
                                    _Float16* __restrict__ out, int total, int cmask) {
    int i = blockIdx.x * blockDim.x + threadIdx.x;
    if (i < total) {
        int c = i & cmask;
        float x = agg[i] + b[c];
        float y = (x - m[c]) * rsqrtf(v[c] + 1e-5f) * g[c] + be[c];
        out[i] = (_Float16)fmaxf(y, 0.0f);
    }
}

__global__ void bias_out_kernel(const float* __restrict__ agg, const float* __restrict__ b,
                                float* __restrict__ out, int total, int cmask) {
    int i = blockIdx.x * blockDim.x + threadIdx.x;
    if (i < total) out[i] = agg[i] + b[i & cmask];
}

// ---------------- WMMA GEMM: C[M,Cout] = A[M,K] * B[Cout,K]^T ----------------
// One wave per 16x16 output tile; K in steps of 32 via v_wmma_f32_16x16x32_f16.
__global__ void gemm_wmma_kernel(const _Float16* __restrict__ A,   // [M x K] row-major
                                 const _Float16* __restrict__ B,   // [Cout x K] row-major (= W)
                                 float* __restrict__ C,            // [M x Cout]
                                 int M, int K, int Cout) {
    int wave = (int)((blockIdx.x * blockDim.x + threadIdx.x) >> 5);
    int lane = threadIdx.x & 31;
    int ntiles = Cout >> 4;
    int mtiles = M >> 4;                 // M is a multiple of 16 here
    if (wave >= mtiles * ntiles) return;
    int mt = wave / ntiles;
    int nt = wave - mt * ntiles;

    int r  = lane & 15;
    int hi = lane >> 4;                  // 0 for lanes 0-15, 1 for lanes 16-31
    int kb = hi * 8;                     // K sub-offset per ISA 16-bit A/B layout

    const _Float16* pa = A + (size_t)(mt * 16 + r) * K + kb;
    const _Float16* pb = B + (size_t)(nt * 16 + r) * K + kb;

    v8f c = {};
#pragma unroll
    for (int k0 = 0; k0 < 128; k0 += 32) {
        v16h a, b;
        ((h8*)&a)[0] = *(const h8*)(pa + k0);
        ((h8*)&a)[1] = *(const h8*)(pa + k0 + 16);
        ((h8*)&b)[0] = *(const h8*)(pb + k0);
        ((h8*)&b)[1] = *(const h8*)(pb + k0 + 16);
        // (neg_a, A, neg_b, B, c_mod, C, reuse_a, reuse_b)
        c = __builtin_amdgcn_wmma_f32_16x16x32_f16(false, a, false, b, (short)0, c, false, false);
    }

    int n    = nt * 16 + r;
    int mrow = mt * 16 + hi * 8;
    float* out = C + (size_t)mrow * Cout + n;
#pragma unroll
    for (int i = 0; i < 8; ++i) out[(size_t)i * Cout] = c[i];
}

// ---------------- host-side launch ----------------

static inline int cdiv(long long a, long long b) { return (int)((a + b - 1) / b); }

extern "C" void kernel_launch(void* const* d_in, const int* in_sizes, int n_in,
                              void* d_out, int out_size, void* d_ws, size_t ws_size,
                              hipStream_t stream) {
    const float* x  = (const float*)d_in[0];
    const int*   ei = (const int*)d_in[1];          // [2, E] flat
    const float* ew = (const float*)d_in[2];
    const float* W1 = (const float*)d_in[3];
    const float* b1 = (const float*)d_in[4];
    const float* g1 = (const float*)d_in[5];
    const float* be1= (const float*)d_in[6];
    const float* m1 = (const float*)d_in[7];
    const float* v1 = (const float*)d_in[8];
    const float* W2 = (const float*)d_in[9];
    const float* b2 = (const float*)d_in[10];
    const float* g2 = (const float*)d_in[11];
    const float* be2= (const float*)d_in[12];
    const float* m2 = (const float*)d_in[13];
    const float* v2 = (const float*)d_in[14];
    const float* W3 = (const float*)d_in[15];
    const float* b3 = (const float*)d_in[16];
    float* out = (float*)d_out;

    const int N = NNODES, E = NEDGES;
    const int* src = ei;
    const int* dst = ei + E;

    // workspace carve-up (256B aligned)
    char* p = (char*)d_ws;
    auto take = [&](size_t bytes) { char* q = p; p += (bytes + 255) & ~(size_t)255; return q; };
    _Float16* xh   = (_Float16*)take((size_t)N * HC * sizeof(_Float16)); // reused as a1/a2 f16
    _Float16* w1h  = (_Float16*)take((size_t)HC * INC * sizeof(_Float16));
    _Float16* w2h  = (_Float16*)take((size_t)HC * HC * sizeof(_Float16));
    _Float16* w3h  = (_Float16*)take((size_t)OUTC * HC * sizeof(_Float16));
    float*    h    = (float*)take((size_t)N * HC * sizeof(float));
    float*    agg  = (float*)take((size_t)N * HC * sizeof(float));
    float*    deg  = (float*)take((size_t)N * sizeof(float));
    float*    dis  = (float*)take((size_t)N * sizeof(float));
    float*    nrm  = (float*)take((size_t)E * sizeof(float));

    const int B256 = 256;

    // 1) f16 conversions
    f32_to_f16_kernel<<<cdiv((long long)N * INC, B256), B256, 0, stream>>>(x, xh, N * INC);
    f32_to_f16_kernel<<<cdiv(HC * INC, B256), B256, 0, stream>>>(W1, w1h, HC * INC);
    f32_to_f16_kernel<<<cdiv(HC * HC, B256), B256, 0, stream>>>(W2, w2h, HC * HC);
    f32_to_f16_kernel<<<cdiv(OUTC * HC, B256), B256, 0, stream>>>(W3, w3h, OUTC * HC);

    // 2) gcn_norm: degree (incl. self-loop), dis = rsqrt(deg), per-edge norm
    deg_init_kernel<<<cdiv(N, B256), B256, 0, stream>>>(deg, N);
    deg_accum_kernel<<<cdiv(E, B256), B256, 0, stream>>>(dst, ew, deg, E);
    dis_kernel<<<cdiv(N, B256), B256, 0, stream>>>(deg, dis, N);
    norm_kernel<<<cdiv(E, B256), B256, 0, stream>>>(src, dst, ew, dis, nrm, E);

    const int gemmBlk = 256;            // 8 waves per block
    long long waves128 = (long long)(N / 16) * (HC / 16);
    long long waves64  = (long long)(N / 16) * (OUTC / 16);

    // ---- layer 1: GCNConv(IN->H) + BN + ReLU ----
    gemm_wmma_kernel<<<cdiv(waves128 * 32, gemmBlk), gemmBlk, 0, stream>>>(xh, w1h, h, N, INC, HC);
    self_init_kernel<<<cdiv((long long)N * HC, B256), B256, 0, stream>>>(h, dis, agg, N * HC, 7);
    edge_scatter_kernel<<<cdiv((long long)E * (HC / 4), B256), B256, 0, stream>>>(
        h, src, dst, nrm, agg, E, HC, 5);
    bias_bn_relu_kernel<<<cdiv((long long)N * HC, B256), B256, 0, stream>>>(
        agg, b1, g1, be1, m1, v1, xh, N * HC, HC - 1);

    // ---- layer 2: GCNConv(H->H) + BN + ReLU ----
    gemm_wmma_kernel<<<cdiv(waves128 * 32, gemmBlk), gemmBlk, 0, stream>>>(xh, w2h, h, N, HC, HC);
    self_init_kernel<<<cdiv((long long)N * HC, B256), B256, 0, stream>>>(h, dis, agg, N * HC, 7);
    edge_scatter_kernel<<<cdiv((long long)E * (HC / 4), B256), B256, 0, stream>>>(
        h, src, dst, nrm, agg, E, HC, 5);
    bias_bn_relu_kernel<<<cdiv((long long)N * HC, B256), B256, 0, stream>>>(
        agg, b2, g2, be2, m2, v2, xh, N * HC, HC - 1);

    // ---- layer 3: GCNConv(H->OUT) ----
    gemm_wmma_kernel<<<cdiv(waves64 * 32, gemmBlk), gemmBlk, 0, stream>>>(xh, w3h, h, N, HC, OUTC);
    self_init_kernel<<<cdiv((long long)N * OUTC, B256), B256, 0, stream>>>(h, dis, agg, N * OUTC, 6);
    edge_scatter_kernel<<<cdiv((long long)E * (OUTC / 4), B256), B256, 0, stream>>>(
        h, src, dst, nrm, agg, E, OUTC, 4);
    bias_out_kernel<<<cdiv((long long)N * OUTC, B256), B256, 0, stream>>>(agg, b3, out, N * OUTC, OUTC - 1);
}
